// CSOCSSC_v115_34454227648703
// MI455X (gfx1250) — matrix operand order repaired and checked by reading, not compiled, not run on hardware
//
#include <hip/hip_runtime.h>
#include <hip/hip_bf16.h>
#include <math.h>

// ---------------------------------------------------------------------------
// MI455X (gfx1250, wave32).
// - All GEMMs: v_wmma_f32_16x16x32_f16, f32 accumulate.
// - Weights + WMMA-feeding activations materialized once as f16 so fragment
//   loads are pure b128 loads (no v_cvt in the inner loops).
// - Distogram head fuses gelu(hi[i]+hj[j]+f1b) into the A fragment (branch-
//   free gelu), stages f16 F2 in LDS (padded rows), 4 N-tiles per wave.
// ---------------------------------------------------------------------------

#define NN   512
#define DD   256
#define DH   32
#define HH   8
#define LL   6
#define BINS 64
#define D4   1024
#define LDSROW 264   // 256 + 8 f16 pad per row (16B) to spread LDS banks

typedef __attribute__((ext_vector_type(16))) _Float16 v16h;
typedef __attribute__((ext_vector_type(8)))  _Float16 v8h;
typedef __attribute__((ext_vector_type(8)))  float    v8f;
typedef __attribute__((ext_vector_type(4)))  float    v4f;

__device__ __forceinline__ float sigmoidf_(float x) {
    return 1.0f / (1.0f + __expf(-x));
}
// Exact-erf gelu for cold epilogues (matches reference nn.GELU default).
__device__ __forceinline__ float gelu_erf(float x) {
    return 0.5f * x * (1.0f + erff(x * 0.70710678118654752f));
}
// Branch-free tanh-form gelu for the 67M-eval distogram hot loop.
__device__ __forceinline__ float gelu_fast(float x) {
    const float u = x * (0.7978845608028654f + 0.0356774081363001f * x * x);
    const float e = __expf(-2.0f * fabsf(u));
    float t = (1.0f - e) / (1.0f + e);
    t = copysignf(t, u);
    return 0.5f * x * (1.0f + t);
}

// ---- WMMA f16 fragment loaders (layouts per CDNA5 ISA 7.12.2, wave32) ------
// A 16x32: lane L row M=L%16; K = base + {0..7} U base+16+{0..7}, base=(L>=16)*8
__device__ __forceinline__ v16h frag_A_h(const _Float16* __restrict__ rowp,
                                         int k0, int lane) {
    const _Float16* p = rowp + k0 + ((lane & 16) ? 8 : 0);
    const v8h lo = *(const v8h*)(p);
    const v8h hi = *(const v8h*)(p + 16);
    v16h a;
#pragma unroll
    for (int t = 0; t < 8; ++t) { a[t] = lo[t]; a[8 + t] = hi[t]; }
    return a;
}
// B 32x16 (=W^T): lane L col N=L%16; K = base + {0..7} U base+16+{0..7}, base=k0+(L>=16)*8
__device__ __forceinline__ v16h frag_B_h(const _Float16* __restrict__ wrow,
                                         int k0, int lane) {
    const _Float16* p = wrow + k0 + ((lane & 16) ? 8 : 0);
    const v8h lo = *(const v8h*)(p);
    const v8h hi = *(const v8h*)(p + 16);
    v16h b;
#pragma unroll
    for (int t = 0; t < 8; ++t) { b[t] = lo[t]; b[8 + t] = hi[t]; }
    return b;
}

// ---- f32 -> f16 conversion (weights, once per launch) ----------------------
__global__ void cvt_f16_kernel(const float* __restrict__ in,
                               _Float16* __restrict__ out, int n) {
    int i = blockIdx.x * 256 + threadIdx.x;
    if (i < n) out[i] = (_Float16)in[i];
}

// ---- Generic fused GEMM: C[M,Nout] = epi(A @ W^T + bias [+ res]) -----------
// f16 operands, f32 accum. One wave computes a 16x64 strip (4 N-tiles reuse
// the A fragment); 8 waves/block. EPI: 0 none, 1 relu, 2 gelu, 3 sigmoid,
// 4 residual-add. Optional f16 mirror output Ch.
template <int EPI>
__global__ void gemm_wmma(const _Float16* __restrict__ A, int lda,
                          const _Float16* __restrict__ W, int ldw,
                          const float* __restrict__ bias,
                          const float* __restrict__ res,
                          float* __restrict__ C, int ldc,
                          _Float16* __restrict__ Ch,
                          int M, int Nout, int K) {
    const int wave = blockIdx.x * 8 + (threadIdx.x >> 5);
    const int lane = threadIdx.x & 31;
    const int ngrp = Nout >> 6;       // groups of 4 n-tiles
    const int tm = wave / ngrp;
    const int tg = wave % ngrp;
    if (tm * 16 >= M) return;         // wave-uniform

    const _Float16* arow = A + (size_t)(tm * 16 + (lane & 15)) * lda;
    const _Float16* wr[4];
#pragma unroll
    for (int nt = 0; nt < 4; ++nt)
        wr[nt] = W + (size_t)(tg * 64 + nt * 16 + (lane & 15)) * ldw;

    v8f acc[4] = {{}, {}, {}, {}};
    for (int k0 = 0; k0 < K; k0 += 32) {
        const v16h a = frag_A_h(arow, k0, lane);
#pragma unroll
        for (int nt = 0; nt < 4; ++nt)
            acc[nt] = __builtin_amdgcn_wmma_f32_16x16x32_f16(
                false, a, false, frag_B_h(wr[nt], k0, lane),
                (short)0, acc[nt], false, false);
    }

    const int mbase = tm * 16 + ((lane & 16) ? 8 : 0);
#pragma unroll
    for (int nt = 0; nt < 4; ++nt) {
        const int n = tg * 64 + nt * 16 + (lane & 15);
        const float bn = bias ? bias[n] : 0.0f;
#pragma unroll
        for (int r = 0; r < 8; ++r) {
            const int m = mbase + r;
            float v = acc[nt][r] + bn;
            if (EPI == 1) v = fmaxf(v, 0.0f);
            else if (EPI == 2) v = gelu_erf(v);
            else if (EPI == 3) v = sigmoidf_(v);
            else if (EPI == 4) v += res[(size_t)m * ldc + n];
            C[(size_t)m * ldc + n] = v;
            if (Ch) Ch[(size_t)m * ldc + n] = (_Float16)v;
        }
    }
}

// ---- Embedding + sinusoidal positional encoding ----------------------------
__global__ void embed_kernel(const int* __restrict__ tokens,
                             const float* __restrict__ emb,
                             float* __restrict__ x,
                             _Float16* __restrict__ xh) {
    const int i = blockIdx.x, d = threadIdx.x;
    const int t = tokens[i];
    const int m = d >> 1;
    const float div = __expf((float)(2 * m) * (-9.210340371976184f / (float)DD));
    const float ang = (float)i * div;
    const float pe = (d & 1) ? cosf(ang) : sinf(ang);
    const float v = emb[t * DD + d] + pe;
    x[i * DD + d] = v;
    xh[i * DD + d] = (_Float16)v;
}

// ---- LayerNorm (one block per row) -----------------------------------------
__global__ void ln_kernel(const float* __restrict__ in,
                          const float* __restrict__ g,
                          const float* __restrict__ b,
                          float* __restrict__ out,
                          _Float16* __restrict__ outh) {
    const int i = blockIdx.x, d = threadIdx.x;
    __shared__ float red[DD];
    const float v = in[i * DD + d];
    red[d] = v; __syncthreads();
    for (int off = 128; off > 0; off >>= 1) {
        if (d < off) red[d] += red[d + off];
        __syncthreads();
    }
    const float mean = red[0] * (1.0f / (float)DD);
    __syncthreads();
    const float dv = v - mean;
    red[d] = dv * dv; __syncthreads();
    for (int off = 128; off > 0; off >>= 1) {
        if (d < off) red[d] += red[d + off];
        __syncthreads();
    }
    const float var = red[0] * (1.0f / (float)DD);
    const float o = dv * rsqrtf(var + 1e-5f) * g[d] + b[d];
    out[i * DD + d] = o;
    outh[i * DD + d] = (_Float16)o;
}

// ---- Attention: one block per (query i, head h); f32 softmax ---------------
__global__ void attn_kernel(const float* __restrict__ qkv,
                            float* __restrict__ attnout,
                            _Float16* __restrict__ attnh) {
    const int i = blockIdx.x, h = blockIdx.y, tid = threadIdx.x;  // block=128
    __shared__ float sc[NN];
    __shared__ float red[128];
    __shared__ float qs[DH];
    if (tid < DH) qs[tid] = qkv[i * (3 * DD) + h * DH + tid];
    __syncthreads();

    float mloc = -1e30f;
    for (int j = tid; j < NN; j += 128) {
        const float* kr = qkv + j * (3 * DD) + DD + h * DH;
        float s = 0.0f;
#pragma unroll
        for (int d = 0; d < DH; ++d) s += qs[d] * kr[d];
        s *= 0.17677669529663687f;  // 1/sqrt(32)
        sc[j] = s;
        mloc = fmaxf(mloc, s);
    }
    red[tid] = mloc; __syncthreads();
    for (int off = 64; off > 0; off >>= 1) {
        if (tid < off) red[tid] = fmaxf(red[tid], red[tid + off]);
        __syncthreads();
    }
    const float mx = red[0]; __syncthreads();

    float sloc = 0.0f;
    for (int j = tid; j < NN; j += 128) {
        const float p = __expf(sc[j] - mx);
        sc[j] = p; sloc += p;
    }
    red[tid] = sloc; __syncthreads();
    for (int off = 64; off > 0; off >>= 1) {
        if (tid < off) red[tid] += red[tid + off];
        __syncthreads();
    }
    const float inv = 1.0f / red[0]; __syncthreads();

    const int d = tid & 31, g = tid >> 5;  // 4 groups x 32 dims
    float acc = 0.0f;
    for (int j = g * 128; j < (g + 1) * 128; ++j)
        acc += sc[j] * qkv[j * (3 * DD) + 2 * DD + h * DH + d];
    red[tid] = acc; __syncthreads();
    if (g == 0) {
        const float o = (red[d] + red[d + 32] + red[d + 64] + red[d + 96]) * inv;
        attnout[i * DD + h * DH + d] = o;
        attnh[i * DD + h * DH + d] = (_Float16)o;
    }
}

// ---- Alpha head part 2 -----------------------------------------------------
__global__ void alpha_kernel(const float* __restrict__ hA,
                             const float* __restrict__ A2,
                             const float* __restrict__ a2b,
                             float* __restrict__ alpha) {
    const int i = blockIdx.x * blockDim.x + threadIdx.x;
    if (i >= NN) return;
    float s = a2b[0];
    for (int k = 0; k < DD; ++k) s += hA[i * DD + k] * A2[k];
    alpha[i] = 1.2f + (3.5f - 1.2f) * sigmoidf_(s);
}

// ---- CSOCRG kernel mix -----------------------------------------------------
__global__ void kern_apply(const float* __restrict__ alpha,
                           const float* __restrict__ lin,
                           float* __restrict__ lout,
                           _Float16* __restrict__ louth,
                           float* __restrict__ lout2) {
    const int i = blockIdx.x, tid = threadIdx.x;  // block=256
    __shared__ float kc[NN];
    __shared__ float red[256];
    const float ai = alpha[i];
    float sl = 0.0f;
    for (int j = tid; j < NN; j += 256) {
        const float r  = fabsf((float)(i - j)) + 1e-4f;
        const float ap = 0.5f * (ai + alpha[j]);
        const float kv = __expf(-ap * __logf(r) - r * (1.0f / 30.0f));
        kc[j] = kv; sl += kv;
    }
    red[tid] = sl; __syncthreads();
    for (int off = 128; off > 0; off >>= 1) {
        if (tid < off) red[tid] += red[tid + off];
        __syncthreads();
    }
    const float inv = 1.0f / (red[0] + 1e-8f);
    __syncthreads();
    float acc = 0.0f;
    for (int j = 0; j < NN; ++j) acc += kc[j] * lin[j * DD + tid];
    const float v = acc * inv;
    lout[i * DD + tid] = v;
    louth[i * DD + tid] = (_Float16)v;
    lout2[i * DD + tid] = v;
}

// ---- Distogram: fused gelu(hi+hj+f1b) -> WMMA vs F2^T (F2 f16 in LDS) ------
__global__ void disto_kernel(const float* __restrict__ hi,
                             const float* __restrict__ hj,
                             const float* __restrict__ f1b,
                             const _Float16* __restrict__ F2h,
                             const float* __restrict__ f2b,
                             float* __restrict__ out) {
    __shared__ _Float16 f2s[BINS * LDSROW];
    // Stage F2 (64x256 f16 = 32KB) into LDS once per block, padded rows.
    for (int idx = threadIdx.x; idx < BINS * 16; idx += 256) {
        const int rowi = idx >> 4;
        const int seg  = idx & 15;  // 16 f16 per segment
        const v8h* src = (const v8h*)(F2h + rowi * DD + seg * 16);
        *(v8h*)(&f2s[rowi * LDSROW + seg * 16])     = src[0];
        *(v8h*)(&f2s[rowi * LDSROW + seg * 16 + 8]) = src[1];
    }
    __syncthreads();

    const int wave = blockIdx.x * 8 + (threadIdx.x >> 5);
    const int lane = threadIdx.x & 31;
    const int i  = wave >> 5;   // wave / (N/16)
    const int jt = wave & 31;   // wave % (N/16)
    const float* hir = hi + (size_t)i * DD;
    const float* hjr = hj + (size_t)(jt * 16 + (lane & 15)) * DD;  // A row = j
    const int kb = (lane & 16) ? 8 : 0;
    const int ksel = (lane & 16) ? 8 : 0;

    v8f acc[4] = {{}, {}, {}, {}};
    for (int k0 = 0; k0 < DD; k0 += 32) {
        __builtin_prefetch(hjr + k0 + 64, 0, 1);
        const float* pi = hir + k0 + kb;
        const float* pj = hjr + k0 + kb;
        const float* pf = f1b + k0 + kb;
        const v4f i0 = *(const v4f*)(pi),      i1 = *(const v4f*)(pi + 4);
        const v4f i2 = *(const v4f*)(pi + 16), i3 = *(const v4f*)(pi + 20);
        const v4f j0 = *(const v4f*)(pj),      j1 = *(const v4f*)(pj + 4);
        const v4f j2 = *(const v4f*)(pj + 16), j3 = *(const v4f*)(pj + 20);
        const v4f f0 = *(const v4f*)(pf),      f1 = *(const v4f*)(pf + 4);
        const v4f f2 = *(const v4f*)(pf + 16), f3 = *(const v4f*)(pf + 20);
        v16h a;
#pragma unroll
        for (int t = 0; t < 4; ++t) {
            a[t]      = (_Float16)gelu_fast(i0[t] + j0[t] + f0[t]);
            a[4 + t]  = (_Float16)gelu_fast(i1[t] + j1[t] + f1[t]);
            a[8 + t]  = (_Float16)gelu_fast(i2[t] + j2[t] + f2[t]);
            a[12 + t] = (_Float16)gelu_fast(i3[t] + j3[t] + f3[t]);
        }
#pragma unroll
        for (int nt = 0; nt < 4; ++nt) {
            const _Float16* p = &f2s[(nt * 16 + (lane & 15)) * LDSROW + k0 + ksel];
            const v8h lo = *(const v8h*)(p);
            const v8h hi2 = *(const v8h*)(p + 16);
            v16h b;
#pragma unroll
            for (int t = 0; t < 8; ++t) { b[t] = lo[t]; b[8 + t] = hi2[t]; }
            acc[nt] = __builtin_amdgcn_wmma_f32_16x16x32_f16(
                false, a, false, b, (short)0, acc[nt], false, false);
        }
    }

    const int jj = jt * 16 + ((lane & 16) ? 8 : 0);
#pragma unroll
    for (int nt = 0; nt < 4; ++nt) {
        const int bin = nt * 16 + (lane & 15);
        const float fb = f2b[bin];
#pragma unroll
        for (int r = 0; r < 8; ++r)
            out[((size_t)i * NN + (jj + r)) * BINS + bin] = acc[nt][r] + fb;
    }
}

// ---- Torsion head part 2 ---------------------------------------------------
__global__ void torsion_kernel(const float* __restrict__ gT,
                               const float* __restrict__ T2,
                               const float* __restrict__ t2b,
                               float* __restrict__ phi,
                               float* __restrict__ psi) {
    const int i = blockIdx.x * blockDim.x + threadIdx.x;
    if (i >= NN) return;
    float s0 = t2b[0], s1 = t2b[1];
    for (int k = 0; k < DD; ++k) {
        const float gv = gT[i * DD + k];
        s0 += gv * T2[k];
        s1 += gv * T2[DD + k];
    }
    phi[i] = tanhf(s0) * 3.14159265358979323846f;
    psi[i] = tanhf(s1) * 3.14159265358979323846f;
}

// ---- Cp projection + diffusion init loop -----------------------------------
__global__ void coords_kernel(const float* __restrict__ lat,
                              const float* __restrict__ Cp,
                              const float* __restrict__ cpb,
                              const float* __restrict__ noise,
                              float* __restrict__ xc) {
    const int i = blockIdx.x * blockDim.x + threadIdx.x;
    if (i >= NN) return;
    float tgt[3];
#pragma unroll
    for (int c = 0; c < 3; ++c) {
        float s = cpb[c];
        for (int k = 0; k < DD; ++k) s += lat[i * DD + k] * Cp[c * DD + k];
        tgt[c] = s;
    }
#pragma unroll
    for (int c = 0; c < 3; ++c) {
        float v = noise[i * 3 + c];
        for (int tt = 7; tt >= 0; --tt) {
            const float a = (float)(tt + 1) * (1.0f / 8.0f);
            v = a * v + (1.0f - a) * tgt[c];
        }
        xc[i * 3 + c] = v;
    }
}

// ---------------------------------------------------------------------------
static inline void cvt(const float* in, _Float16* out, int n, hipStream_t s) {
    cvt_f16_kernel<<<(n + 255) / 256, 256, 0, s>>>(in, out, n);
}

extern "C" void kernel_launch(void* const* d_in, const int* in_sizes, int n_in,
                              void* d_out, int out_size, void* d_ws, size_t ws_size,
                              hipStream_t stream) {
    (void)in_sizes; (void)n_in; (void)out_size; (void)ws_size;
    const int*   tokens = (const int*)  d_in[0];
    const float* emb    = (const float*)d_in[1];
    const float* Wqkv   = (const float*)d_in[2];
    const float* bqkv   = (const float*)d_in[3];
    const float* Wo     = (const float*)d_in[4];
    const float* bo     = (const float*)d_in[5];
    const float* g1     = (const float*)d_in[6];
    const float* be1    = (const float*)d_in[7];
    const float* W1     = (const float*)d_in[8];
    const float* b1     = (const float*)d_in[9];
    const float* W2     = (const float*)d_in[10];
    const float* b2     = (const float*)d_in[11];
    const float* g2     = (const float*)d_in[12];
    const float* be2    = (const float*)d_in[13];
    const float* A1     = (const float*)d_in[14];
    const float* a1b    = (const float*)d_in[15];
    const float* A2     = (const float*)d_in[16];
    const float* a2b    = (const float*)d_in[17];
    const float* Cf     = (const float*)d_in[18];
    const float* cb     = (const float*)d_in[19];
    const float* F1     = (const float*)d_in[20];
    const float* f1b    = (const float*)d_in[21];
    const float* F2     = (const float*)d_in[22];
    const float* f2b    = (const float*)d_in[23];
    const float* T1     = (const float*)d_in[24];
    const float* t1b    = (const float*)d_in[25];
    const float* T2     = (const float*)d_in[26];
    const float* t2b    = (const float*)d_in[27];
    const float* Cp     = (const float*)d_in[28];
    const float* cpb    = (const float*)d_in[29];
    const float* noise  = (const float*)d_in[30];

    // Output regions (flat, return order)
    float* out       = (float*)d_out;
    float* o_latent  = out;
    float* o_alpha   = out + 131072;     // N*D
    float* o_contact = out + 131584;     // + N
    float* o_disto   = out + 393728;     // + N*N
    float* o_phi     = out + 17170944;   // + N*N*BINS
    float* o_psi     = out + 17171456;
    float* o_xc      = out + 17171968;

    // Workspace: f32 region, then f16 region
    float* ws   = (float*)d_ws;
    float* xbuf = ws;                  // N*D
    float* tmp  = xbuf + NN * DD;      // N*D
    float* qkv  = tmp  + NN * DD;      // N*3D
    float* attn = qkv  + NN * 3 * DD;  // N*D  (also hA / gT scratch)
    float* ff   = attn + NN * DD;      // N*4D
    float* lat2 = ff   + NN * 4 * DD;  // N*D
    float* hh   = lat2 + NN * DD;      // N*D  (f32 scratch, unused after mirror)
    float* hi   = hh   + NN * DD;      // N*D
    float* hj   = hi   + NN * DD;      // N*D

    _Float16* h16     = (_Float16*)(hj + NN * DD);
    _Float16* xh      = h16;                      // N*D
    _Float16* attn_h  = xh     + NN * DD;         // N*D
    _Float16* ff_h    = attn_h + NN * DD;         // N*4D
    _Float16* lat2_h  = ff_h   + NN * D4;         // N*D
    _Float16* hh_h    = lat2_h + NN * DD;         // N*D
    _Float16* Wqkv_h  = hh_h   + NN * DD;         // L*3D*D
    _Float16* Wo_h    = Wqkv_h + LL * 3 * DD * DD;
    _Float16* W1_h    = Wo_h   + LL * DD * DD;
    _Float16* W2_h    = W1_h   + LL * D4 * DD;
    _Float16* A1_h    = W2_h   + LL * DD * D4;
    _Float16* Cf_h    = A1_h   + DD * DD;
    _Float16* F1_h    = Cf_h   + DD * DD;         // D x 2D
    _Float16* F2_h    = F1_h   + DD * 2 * DD;     // BINS x D
    _Float16* T1_h    = F2_h   + BINS * DD;

    // One-time (per launch) weight conversions to f16
    cvt(Wqkv, Wqkv_h, LL * 3 * DD * DD, stream);
    cvt(Wo,   Wo_h,   LL * DD * DD,     stream);
    cvt(W1,   W1_h,   LL * D4 * DD,     stream);
    cvt(W2,   W2_h,   LL * DD * D4,     stream);
    cvt(A1,   A1_h,   DD * DD,          stream);
    cvt(Cf,   Cf_h,   DD * DD,          stream);
    cvt(F1,   F1_h,   DD * 2 * DD,      stream);
    cvt(F2,   F2_h,   BINS * DD,        stream);
    cvt(T1,   T1_h,   DD * DD,          stream);

    embed_kernel<<<NN, DD, 0, stream>>>(tokens, emb, xbuf, xh);

    for (int l = 0; l < LL; ++l) {
        // qkv = x @ Wqkv^T + bqkv      [512,768]; waves = 32*12 -> 48 blocks
        gemm_wmma<0><<<48, 256, 0, stream>>>(
            xh, DD, Wqkv_h + (size_t)l * 3 * DD * DD, DD,
            bqkv + l * 3 * DD, nullptr, qkv, 3 * DD, nullptr, NN, 3 * DD, DD);
        attn_kernel<<<dim3(NN, HH), 128, 0, stream>>>(qkv, attn, attn_h);
        // tmp = x + attn @ Wo^T + bo ; x = LN(tmp)
        gemm_wmma<4><<<16, 256, 0, stream>>>(
            attn_h, DD, Wo_h + (size_t)l * DD * DD, DD,
            bo + l * DD, xbuf, tmp, DD, nullptr, NN, DD, DD);
        ln_kernel<<<NN, DD, 0, stream>>>(tmp, g1 + l * DD, be1 + l * DD, xbuf, xh);
        // ff = relu(x @ W1^T + b1)     [512,1024]; 32*16 waves -> 64 blocks
        gemm_wmma<1><<<64, 256, 0, stream>>>(
            xh, DD, W1_h + (size_t)l * D4 * DD, DD,
            b1 + l * D4, nullptr, ff, D4, ff_h, NN, D4, DD);
        // tmp = x + ff @ W2^T + b2 ; x = LN(tmp)
        gemm_wmma<4><<<16, 256, 0, stream>>>(
            ff_h, D4, W2_h + (size_t)l * DD * D4, D4,
            b2 + l * DD, xbuf, tmp, DD, nullptr, NN, DD, D4);
        ln_kernel<<<NN, DD, 0, stream>>>(tmp, g2 + l * DD, be2 + l * DD, xbuf, xh);
    }

    // Alpha head (pre-kern latent = xbuf)
    gemm_wmma<2><<<16, 256, 0, stream>>>(
        xh, DD, A1_h, DD, a1b, nullptr, attn, DD, nullptr, NN, DD, DD);
    alpha_kernel<<<2, 256, 0, stream>>>(attn, A2, a2b, o_alpha);

    // CSOCRG mixing -> lat2 (+f16 mirror) and latent output
    kern_apply<<<NN, 256, 0, stream>>>(o_alpha, xbuf, lat2, lat2_h, o_latent);

    // Contact head
    gemm_wmma<0><<<16, 256, 0, stream>>>(
        lat2_h, DD, Cf_h, DD, cb, nullptr, hh, DD, hh_h, NN, DD, DD);
    gemm_wmma<3><<<32, 256, 0, stream>>>(
        hh_h, DD, hh_h, DD, nullptr, nullptr, o_contact, NN, nullptr, NN, NN, DD);

    // Distogram: hi = lat2 @ F1[:,:D]^T, hj = lat2 @ F1[:,D:]^T (bias fused later)
    gemm_wmma<0><<<16, 256, 0, stream>>>(
        lat2_h, DD, F1_h, 2 * DD, nullptr, nullptr, hi, DD, nullptr, NN, DD, DD);
    gemm_wmma<0><<<16, 256, 0, stream>>>(
        lat2_h, DD, F1_h + DD, 2 * DD, nullptr, nullptr, hj, DD, nullptr, NN, DD, DD);
    disto_kernel<<<2048, 256, 0, stream>>>(hi, hj, f1b, F2_h, f2b, o_disto);

    // Torsion head
    gemm_wmma<2><<<16, 256, 0, stream>>>(
        lat2_h, DD, T1_h, DD, t1b, nullptr, attn, DD, nullptr, NN, DD, DD);
    torsion_kernel<<<2, 256, 0, stream>>>(attn, T2, t2b, o_phi, o_psi);

    // Coordinates
    coords_kernel<<<2, 256, 0, stream>>>(lat2, Cp, cpb, noise, o_xc);
}
